// RSSM_6665789243871
// MI455X (gfx1250) — matrix operand order repaired and checked by reading, not compiled, not run on hardware
//
#include <hip/hip_runtime.h>

// ---------------------------------------------------------------------------
// RSSM step for MI455X (gfx1250): bf16 WMMA GEMMs with f32 accumulation.
// Weights are pre-swizzled into WMMA-fragment tile order so all LDS staging
// is contiguous b128 traffic. Staging uses GLOBAL_LOAD_ASYNC_TO_LDS_B128
// (ASYNCcnt) to avoid the VGPR round trip and the mid-loop loadcnt stall.
// ---------------------------------------------------------------------------

typedef __bf16 bf16_t;
typedef __attribute__((ext_vector_type(16))) __bf16 v16bf;
typedef __attribute__((ext_vector_type(8)))  __bf16 v8bf;
typedef __attribute__((ext_vector_type(8)))  float  v8f;
typedef __attribute__((ext_vector_type(4)))  int    v4i;

#define B_   8192
#define S_   1024
#define A_   32
#define O_   2048
#define H_   1024
#define L_   256
#define IN1  (S_ + A_ + O_)   // 3104 = 97*32
#define GIN  (L_ + A_)        // 288  = 9*32

#if __has_builtin(__builtin_amdgcn_global_load_async_to_lds_b128) && \
    __has_builtin(__builtin_amdgcn_s_wait_asynccnt)
#define USE_ASYNC 1
#else
#define USE_ASYNC 0
#endif

typedef __attribute__((address_space(1))) v4i as1_v4i;  // global
typedef __attribute__((address_space(3))) v4i as3_v4i;  // LDS

// 16-byte memory->LDS copy (async when available).
__device__ __forceinline__ void copy16(bf16_t* lds_dst, const bf16_t* g_src) {
#if USE_ASYNC
  __builtin_amdgcn_global_load_async_to_lds_b128(
      (as1_v4i*)g_src, (as3_v4i*)lds_dst, /*offset=*/0, /*cpol=*/0);
#else
  *(v8bf*)lds_dst = *(const v8bf*)g_src;
#endif
}

// Wait for this wave's outstanding staging copies (before the barrier).
__device__ __forceinline__ void stage_wait() {
#if USE_ASYNC
  __builtin_amdgcn_s_wait_asynccnt(0);
#endif
}

// ---------------- WMMA helper ----------------
__device__ __forceinline__ v8f wmma_bf16(v16bf a, v16bf b, v8f c) {
  return __builtin_amdgcn_wmma_f32_16x16x32_bf16(
      /*neg_a=*/false, a, /*neg_b=*/false, b,
      /*c_mod=*/(short)0, c, /*reuse_a=*/false, /*reuse_b=*/false);
}

__device__ __forceinline__ v8f vzero8() {
  v8f z = {0.f, 0.f, 0.f, 0.f, 0.f, 0.f, 0.f, 0.f};
  return z;
}

// Swizzled weight layout: idx(k, n) = ((k/32)*N + n)*32 + (k%32).
// A 32xN k-slab is contiguous [n][k] — exactly the LDS B-tile layout.
__device__ __forceinline__ size_t swz_idx(int k, long n, int N) {
  return ((size_t)(k >> 5) * N + n) * 32 + (k & 31);
}

// ---------------- LDS staging (all contiguous 16B chunks) ----------------
// A tile in LDS: [rows][32] bf16, row-major (row stride 32).
__device__ __forceinline__ void stage_a(bf16_t* lA, const bf16_t* __restrict__ g,
                                        long m0, int k0, int lda, int rows, int t) {
  const int chunks = rows * 4;            // one 16B chunk each
  for (int i = t; i < chunks; i += 256) {
    int r  = i >> 2;
    int kk = (i & 3) << 3;
    copy16(lA + r * 32 + kk, g + (m0 + r) * (long)lda + k0 + kk);
  }
}

// B tile in LDS: [n][k] (n stride 32). Source is pre-swizzled, so this is a
// straight contiguous 8KB copy.
__device__ __forceinline__ void stage_b(bf16_t* lB, const bf16_t* __restrict__ g,
                                        int kt, long n0, int N, int t) {
  const bf16_t* src = g + ((size_t)kt * N + n0) * 32;
  for (int i = t; i < 128 * 4; i += 256) {
    copy16(lB + i * 8, src + i * 8);
  }
}

// ---------------- fragment loads (ISA 7.12.2 layouts) ----------------
// A 16x32 bf16: lane<16 holds row=lane, K={0..7,16..23}; lane>=16 row=lane-16,
// K={8..15,24..31}.
__device__ __forceinline__ v16bf load_a_frag(const bf16_t* lA, int m0, int lane) {
  const bf16_t* base = lA + (m0 + (lane & 15)) * 32;
  int off0 = (lane < 16) ? 0 : 8;
  union { v16bf v; v8bf h[2]; } u;
  u.h[0] = *(const v8bf*)(base + off0);
  u.h[1] = *(const v8bf*)(base + off0 + 16);
  return u.v;
}

// B 32x16 bf16: lanes 0..15 hold col=lane, K={0..7,16..23}; lanes 16..31 K={8..15,24..31}.
__device__ __forceinline__ v16bf load_b_frag(const bf16_t* lB, int n0, int lane) {
  const bf16_t* base = lB + (n0 + (lane & 15)) * 32 + ((lane < 16) ? 0 : 8);
  union { v16bf v; v8bf h[2]; } u;
  u.h[0] = *(const v8bf*)(base);
  u.h[1] = *(const v8bf*)(base + 16);
  return u.v;
}

// ---------------- weight conversion into swizzled bf16 ----------------
// Logical B-matrix is (K x N). TRANS=0: src is K x N row-major (rep_W1/W2).
// TRANS=1: src is N x K row-major (gru_Wih/Whh, torch [out,in] layout).
template <int TRANS>
__global__ void cvt_swz_kernel(const float* __restrict__ src, bf16_t* __restrict__ dst,
                               int K, int N) {
  long i = (long)blockIdx.x * blockDim.x + threadIdx.x;
  if (i >= (long)K * N) return;
  int k = (int)(i / N), n = (int)(i % N);
  float v = TRANS ? src[(long)n * K + k] : src[(long)k * N + n];
  dst[swz_idx(k, n, N)] = (bf16_t)v;
}

// xcat = bf16 concat([prev_state, prev_action, observation]) : B x 3104
// 8 elements per thread; section boundaries (1024, 1056) are 8-aligned.
__global__ void pack_x_kernel(const float* __restrict__ ps, const float* __restrict__ pa,
                              const float* __restrict__ ob, bf16_t* __restrict__ xcat) {
  long i = (long)blockIdx.x * blockDim.x + threadIdx.x;
  if (i >= (long)B_ * (IN1 / 8)) return;
  int m = (int)(i / (IN1 / 8));
  int k = (int)(i % (IN1 / 8)) * 8;
  const float* src;
  if (k < S_)            src = ps + (long)m * S_ + k;
  else if (k < S_ + A_)  src = pa + (long)m * A_ + (k - S_);
  else                   src = ob + (long)m * O_ + (k - S_ - A_);
  float4 a = *(const float4*)src;
  float4 b = *(const float4*)(src + 4);
  v8bf o;
  o[0] = (bf16_t)a.x; o[1] = (bf16_t)a.y; o[2] = (bf16_t)a.z; o[3] = (bf16_t)a.w;
  o[4] = (bf16_t)b.x; o[5] = (bf16_t)b.y; o[6] = (bf16_t)b.z; o[7] = (bf16_t)b.w;
  *(v8bf*)(xcat + i * 8) = o;
}

// sample = mean + sqrt(std)*eps ; xg = bf16([sample | prev_action]) : B x 288
__global__ void sample_pack_kernel(const float* __restrict__ meanI, const float* __restrict__ stdI,
                                   const float* __restrict__ eps, const float* __restrict__ pa,
                                   float* __restrict__ sampleO, bf16_t* __restrict__ xg) {
  long i = (long)blockIdx.x * blockDim.x + threadIdx.x;
  if (i >= (long)B_ * GIN) return;
  int m = (int)(i / GIN), j = (int)(i % GIN);
  if (j < L_) {
    long idx = (long)m * L_ + j;
    float s  = meanI[idx] + sqrtf(stdI[idx]) * eps[idx];
    sampleO[idx] = s;
    xg[i] = (bf16_t)s;
  } else {
    xg[i] = (bf16_t)pa[(long)m * A_ + (j - L_)];
  }
}

// ---------------- generic 128x128 bf16 GEMM ----------------
// MODE 0: C = relu(A*W + bias) -> bf16 Cbf          (GEMM1)
// MODE 1: A*W + bias -> f32 split mean/std halves   (GEMM2)
template <int MODE>
__global__ __launch_bounds__(256) void gemm128_kernel(
    const bf16_t* __restrict__ A, int lda,
    const bf16_t* __restrict__ Wswz, int N, int K,
    const float* __restrict__ bias,
    bf16_t* __restrict__ Cbf, int ldc,
    float* __restrict__ meanOut, float* __restrict__ stdOut) {
  __shared__ bf16_t lA[2][128 * 32];
  __shared__ bf16_t lB[2][128 * 32];
  const int  t    = threadIdx.x;
  const int  lane = t & 31;
  const int  w    = t >> 5;
  const int  wm   = (w >> 2) * 64;  // wave grid 2(M) x 4(N)
  const int  wn   = (w & 3) * 32;
  const long m0   = (long)blockIdx.y * 128;
  const long n0   = (long)blockIdx.x * 128;

  v8f acc[4][2];
#pragma unroll
  for (int mi = 0; mi < 4; ++mi)
#pragma unroll
    for (int ni = 0; ni < 2; ++ni) acc[mi][ni] = vzero8();

  const int kt_n = K >> 5;
  stage_a(lA[0], A, m0, 0, lda, 128, t);
  stage_b(lB[0], Wswz, 0, n0, N, t);
  stage_wait();
  __syncthreads();

  for (int kt = 0; kt < kt_n; ++kt) {
    const int cur = kt & 1;
    if (kt + 1 < kt_n) {
      stage_a(lA[cur ^ 1], A, m0, (kt + 1) << 5, lda, 128, t);
      stage_b(lB[cur ^ 1], Wswz, kt + 1, n0, N, t);
    }
    if (kt + 2 < kt_n) {  // global_prefetch_b8 for the tile after next
      __builtin_prefetch(A + (m0 + (t >> 1)) * (long)lda + ((long)(kt + 2) << 5), 0, 1);
      __builtin_prefetch(Wswz + ((size_t)(kt + 2) * N + n0) * 32 + (size_t)t * 16, 0, 1);
    }
    v16bf af[4], bfr[2];
#pragma unroll
    for (int mi = 0; mi < 4; ++mi) af[mi] = load_a_frag(lA[cur], wm + mi * 16, lane);
#pragma unroll
    for (int ni = 0; ni < 2; ++ni) bfr[ni] = load_b_frag(lB[cur], wn + ni * 16, lane);
#pragma unroll
    for (int mi = 0; mi < 4; ++mi)
#pragma unroll
      for (int ni = 0; ni < 2; ++ni)
        acc[mi][ni] = wmma_bf16(af[mi], bfr[ni], acc[mi][ni]);
    stage_wait();
    __syncthreads();
  }

  // Epilogue. C layout: lane<16 -> (M = sub+e,   N = lane)
  //                     lane>=16-> (M = sub+e+8, N = lane-16)
#pragma unroll
  for (int mi = 0; mi < 4; ++mi)
#pragma unroll
    for (int ni = 0; ni < 2; ++ni) {
      const long col = n0 + wn + ni * 16 + (lane & 15);
      const float bv = bias[col];
#pragma unroll
      for (int e = 0; e < 8; ++e) {
        const long row = m0 + wm + mi * 16 + ((lane < 16) ? e : e + 8);
        float v = acc[mi][ni][e] + bv;
        if (MODE == 0) {
          v = v > 0.f ? v : 0.f;
          Cbf[row * ldc + col] = (bf16_t)v;
        } else {
          if (col < L_) meanOut[row * L_ + col] = v;
          else          stdOut[row * L_ + (col - L_)] = v;
        }
      }
    }
}

// ---------------- fused GRU kernel ----------------
// Computes r,z,gi_n,gh_n accumulators over two K-loops (xg*WihT, state*WhhT),
// then applies GRU gating in-register. Block tile 64(M) x 128(N of H).
__global__ __launch_bounds__(256) void gru_fused_kernel(
    const bf16_t* __restrict__ xg,    // B x 288 bf16
    const bf16_t* __restrict__ st,    // prev_state bf16 = xcat rows, lda=IN1
    const bf16_t* __restrict__ wihS,  // swizzled (288 x 3072)
    const bf16_t* __restrict__ whhS,  // swizzled (1024 x 3072)
    const float* __restrict__ bih, const float* __restrict__ bhh,
    const float* __restrict__ prev_state,  // f32 B x 1024
    float* __restrict__ belief) {          // f32 B x 1024
  __shared__ bf16_t lA[2][64 * 32];
  __shared__ bf16_t lB[2][3][128 * 32];
  const int  t    = threadIdx.x;
  const int  lane = t & 31;
  const int  w    = t >> 5;
  const int  wm   = (w >> 2) * 32;  // wave grid 2(M) x 4(N), wave tile 32x32
  const int  wn   = (w & 3) * 32;
  const long m0   = (long)blockIdx.y * 64;
  const long n0   = (long)blockIdx.x * 128;
  const int  NW   = 3 * H_;

  v8f accR[2][2], accZ[2][2], accGi[2][2], accGh[2][2];
#pragma unroll
  for (int mi = 0; mi < 2; ++mi)
#pragma unroll
    for (int ni = 0; ni < 2; ++ni) {
      accR[mi][ni] = vzero8(); accZ[mi][ni] = vzero8();
      accGi[mi][ni] = vzero8(); accGh[mi][ni] = vzero8();
    }

  // ---- Phase 1: gi = xg @ WihT (K = 288) -> accR, accZ, accGi ----
  stage_a(lA[0], xg, m0, 0, GIN, 64, t);
#pragma unroll
  for (int s = 0; s < 3; ++s) stage_b(lB[0][s], wihS, 0, (long)s * H_ + n0, NW, t);
  stage_wait();
  __syncthreads();
  for (int kt = 0; kt < GIN / 32; ++kt) {
    const int cur = kt & 1;
    if (kt + 1 < GIN / 32) {
      stage_a(lA[cur ^ 1], xg, m0, (kt + 1) << 5, GIN, 64, t);
#pragma unroll
      for (int s = 0; s < 3; ++s)
        stage_b(lB[cur ^ 1][s], wihS, kt + 1, (long)s * H_ + n0, NW, t);
    }
    v16bf af0 = load_a_frag(lA[cur], wm, lane);
    v16bf af1 = load_a_frag(lA[cur], wm + 16, lane);
    {
      v16bf b0 = load_b_frag(lB[cur][0], wn, lane), b1 = load_b_frag(lB[cur][0], wn + 16, lane);
      accR[0][0] = wmma_bf16(af0, b0, accR[0][0]); accR[0][1] = wmma_bf16(af0, b1, accR[0][1]);
      accR[1][0] = wmma_bf16(af1, b0, accR[1][0]); accR[1][1] = wmma_bf16(af1, b1, accR[1][1]);
    }
    {
      v16bf b0 = load_b_frag(lB[cur][1], wn, lane), b1 = load_b_frag(lB[cur][1], wn + 16, lane);
      accZ[0][0] = wmma_bf16(af0, b0, accZ[0][0]); accZ[0][1] = wmma_bf16(af0, b1, accZ[0][1]);
      accZ[1][0] = wmma_bf16(af1, b0, accZ[1][0]); accZ[1][1] = wmma_bf16(af1, b1, accZ[1][1]);
    }
    {
      v16bf b0 = load_b_frag(lB[cur][2], wn, lane), b1 = load_b_frag(lB[cur][2], wn + 16, lane);
      accGi[0][0] = wmma_bf16(af0, b0, accGi[0][0]); accGi[0][1] = wmma_bf16(af0, b1, accGi[0][1]);
      accGi[1][0] = wmma_bf16(af1, b0, accGi[1][0]); accGi[1][1] = wmma_bf16(af1, b1, accGi[1][1]);
    }
    stage_wait();
    __syncthreads();
  }

  // ---- Phase 2: gh = prev_state @ WhhT (K = 1024) -> accR, accZ, accGh ----
  stage_a(lA[0], st, m0, 0, IN1, 64, t);
#pragma unroll
  for (int s = 0; s < 3; ++s) stage_b(lB[0][s], whhS, 0, (long)s * H_ + n0, NW, t);
  stage_wait();
  __syncthreads();
  for (int kt = 0; kt < H_ / 32; ++kt) {
    const int cur = kt & 1;
    if (kt + 1 < H_ / 32) {
      stage_a(lA[cur ^ 1], st, m0, (kt + 1) << 5, IN1, 64, t);
#pragma unroll
      for (int s = 0; s < 3; ++s)
        stage_b(lB[cur ^ 1][s], whhS, kt + 1, (long)s * H_ + n0, NW, t);
    }
    if (kt + 2 < H_ / 32) {
      __builtin_prefetch(st + (m0 + (t >> 2)) * (long)IN1 + ((long)(kt + 2) << 5), 0, 1);
      __builtin_prefetch(whhS + ((size_t)(kt + 2) * NW + n0) * 32 + (size_t)t * 16, 0, 1);
    }
    v16bf af0 = load_a_frag(lA[cur], wm, lane);
    v16bf af1 = load_a_frag(lA[cur], wm + 16, lane);
    {
      v16bf b0 = load_b_frag(lB[cur][0], wn, lane), b1 = load_b_frag(lB[cur][0], wn + 16, lane);
      accR[0][0] = wmma_bf16(af0, b0, accR[0][0]); accR[0][1] = wmma_bf16(af0, b1, accR[0][1]);
      accR[1][0] = wmma_bf16(af1, b0, accR[1][0]); accR[1][1] = wmma_bf16(af1, b1, accR[1][1]);
    }
    {
      v16bf b0 = load_b_frag(lB[cur][1], wn, lane), b1 = load_b_frag(lB[cur][1], wn + 16, lane);
      accZ[0][0] = wmma_bf16(af0, b0, accZ[0][0]); accZ[0][1] = wmma_bf16(af0, b1, accZ[0][1]);
      accZ[1][0] = wmma_bf16(af1, b0, accZ[1][0]); accZ[1][1] = wmma_bf16(af1, b1, accZ[1][1]);
    }
    {
      v16bf b0 = load_b_frag(lB[cur][2], wn, lane), b1 = load_b_frag(lB[cur][2], wn + 16, lane);
      accGh[0][0] = wmma_bf16(af0, b0, accGh[0][0]); accGh[0][1] = wmma_bf16(af0, b1, accGh[0][1]);
      accGh[1][0] = wmma_bf16(af1, b0, accGh[1][0]); accGh[1][1] = wmma_bf16(af1, b1, accGh[1][1]);
    }
    stage_wait();
    __syncthreads();
  }

  // ---- Gating epilogue ----
#pragma unroll
  for (int mi = 0; mi < 2; ++mi)
#pragma unroll
    for (int ni = 0; ni < 2; ++ni) {
      const long col  = n0 + wn + ni * 16 + (lane & 15);   // 0..H-1
      const float brI = bih[col],          brH = bhh[col];
      const float bzI = bih[H_ + col],     bzH = bhh[H_ + col];
      const float bnI = bih[2 * H_ + col], bnH = bhh[2 * H_ + col];
#pragma unroll
      for (int e = 0; e < 8; ++e) {
        const long row = m0 + wm + mi * 16 + ((lane < 16) ? e : e + 8);
        const float r  = 1.f / (1.f + __expf(-(accR[mi][ni][e] + brI + brH)));
        const float z  = 1.f / (1.f + __expf(-(accZ[mi][ni][e] + bzI + bzH)));
        const float ng = tanhf(accGi[mi][ni][e] + bnI + r * (accGh[mi][ni][e] + bnH));
        const float h  = prev_state[row * H_ + col];
        belief[row * H_ + col] = (1.f - z) * ng + z * h;
      }
    }
}

// ---------------------------------------------------------------------------
extern "C" void kernel_launch(void* const* d_in, const int* in_sizes, int n_in,
                              void* d_out, int out_size, void* d_ws, size_t ws_size,
                              hipStream_t stream) {
  const float* prev_state  = (const float*)d_in[0];
  const float* prev_action = (const float*)d_in[1];
  const float* observation = (const float*)d_in[2];
  const float* eps         = (const float*)d_in[3];
  const float* rep_W1      = (const float*)d_in[4];
  const float* rep_b1      = (const float*)d_in[5];
  const float* rep_W2      = (const float*)d_in[6];
  const float* rep_b2      = (const float*)d_in[7];
  const float* gru_Wih     = (const float*)d_in[8];
  const float* gru_Whh     = (const float*)d_in[9];
  const float* gru_bih     = (const float*)d_in[10];
  const float* gru_bhh     = (const float*)d_in[11];

  float* out       = (float*)d_out;
  float* meanOut   = out;
  float* stdOut    = out + (long)B_ * L_;
  float* sampleOut = out + (long)2 * B_ * L_;
  float* beliefOut = out + (long)3 * B_ * L_;

  // workspace layout (bf16, all 16B-aligned by construction) ~ 88 MB
  bf16_t* xcat = (bf16_t*)d_ws;                 // B x 3104
  bf16_t* w1b  = xcat + (size_t)B_ * IN1;       // swz 3104 x 1024
  bf16_t* w2b  = w1b  + (size_t)IN1 * H_;       // swz 1024 x 512
  bf16_t* wihS = w2b  + (size_t)H_ * (2 * L_);  // swz 288  x 3072
  bf16_t* whhS = wihS + (size_t)GIN * (3 * H_); // swz 1024 x 3072
  bf16_t* h1   = whhS + (size_t)H_ * (3 * H_);  // B x 1024
  bf16_t* xgp  = h1   + (size_t)B_ * H_;        // B x 288

  const int T = 256;
  // weight conversions into swizzled bf16 fragment layout
  cvt_swz_kernel<0><<<(int)(((long)IN1 * H_ + T - 1) / T), T, 0, stream>>>(rep_W1, w1b, IN1, H_);
  cvt_swz_kernel<0><<<(int)(((long)H_ * 2 * L_ + T - 1) / T), T, 0, stream>>>(rep_W2, w2b, H_, 2 * L_);
  cvt_swz_kernel<1><<<(int)(((long)GIN * 3 * H_ + T - 1) / T), T, 0, stream>>>(gru_Wih, wihS, GIN, 3 * H_);
  cvt_swz_kernel<1><<<(int)(((long)H_ * 3 * H_ + T - 1) / T), T, 0, stream>>>(gru_Whh, whhS, H_, 3 * H_);
  // activation packing (16B vectorized)
  pack_x_kernel<<<(int)(((long)B_ * (IN1 / 8) + T - 1) / T), T, 0, stream>>>(
      prev_state, prev_action, observation, xcat);

  // GEMM1: h1 = relu(xcat @ W1 + b1)  [8192 x 1024, K=3104]
  gemm128_kernel<0><<<dim3(H_ / 128, B_ / 128), T, 0, stream>>>(
      xcat, IN1, w1b, H_, IN1, rep_b1, h1, H_, nullptr, nullptr);

  // GEMM2: mean|std = h1 @ W2 + b2    [8192 x 512, K=1024]
  gemm128_kernel<1><<<dim3((2 * L_) / 128, B_ / 128), T, 0, stream>>>(
      h1, H_, w2b, 2 * L_, H_, rep_b2, nullptr, 0, meanOut, stdOut);

  // sample + xg packing
  sample_pack_kernel<<<(int)(((long)B_ * GIN + T - 1) / T), T, 0, stream>>>(
      meanOut, stdOut, eps, prev_action, sampleOut, xgp);

  // fused GRU: belief  [8192 x 1024]
  gru_fused_kernel<<<dim3(H_ / 128, B_ / 64), T, 0, stream>>>(
      xgp, xcat, wihS, whhS, gru_bih, gru_bhh, prev_state, beliefOut);
}
